// MultiHeadAttention_15599321219615
// MI455X (gfx1250) — compile-verified, matrix operand order checked by reading, not run on hardware
//
#include <hip/hip_runtime.h>
#include <math.h>

typedef __bf16 bf16;
typedef __attribute__((ext_vector_type(16))) __bf16 v16bf;
typedef __attribute__((ext_vector_type(8)))  __bf16 v8bf;
typedef __attribute__((ext_vector_type(8)))  float  v8f;

#define BATCH   4
#define C_IN    256
#define N_TOK   2304     // 48*48
#define NHEAD   8
#define DHEAD   64
#define QKVDIM  512      // NHEAD*DHEAD

// ---------------------------------------------------------------------------
// WMMA helpers (gfx1250, wave32). 16-bit A-matrix 16x32 layout:
//   lane = hf*16 + ln ; A row = ln ; elements e0..7 -> K = k0 + hf*8 + e
//                                   elements e8..15 -> K = k0 + 16 + hf*8 + (e-8)
// B fragment mirrors A with lane = output column, reading Bt[N][K] row-major.
// Both load as two contiguous 16-byte chunks per lane.
// ---------------------------------------------------------------------------
__device__ __forceinline__ v16bf load_frag(const bf16* __restrict__ p, int ld,
                                           int row, int k0, int lane) {
  const int hf = (lane >> 4) & 1;
  const bf16* a = p + (size_t)row * ld + k0 + hf * 8;
  v8bf lo = *(const v8bf*)(a);
  v8bf hi = *(const v8bf*)(a + 16);
  v16bf r;
#pragma unroll
  for (int i = 0; i < 8; ++i) { r[i] = lo[i]; r[i + 8] = hi[i]; }
  return r;
}

__device__ __forceinline__ v8f wmma_bf16(v16bf a, v16bf b, v8f c) {
  return __builtin_amdgcn_wmma_f32_16x16x32_bf16(false, a, false, b,
                                                 (short)0, c, false, false);
}

__device__ __forceinline__ void wait_lds() {
  asm volatile("s_wait_dscnt 0" ::: "memory");
}

// ---------------------------------------------------------------------------
// Prep: x[B][C][N] (f32) -> tokens t[B*N][C] (bf16)
// ---------------------------------------------------------------------------
__global__ void prep_tokens(const float* __restrict__ x, bf16* __restrict__ t) {
  int idx = blockIdx.x * 256 + threadIdx.x;            // B*N*C = 2359296 exact
  int c  = idx & (C_IN - 1);
  int bn = idx >> 8;
  int n  = bn % N_TOK;
  int b  = bn / N_TOK;
  t[idx] = (bf16)x[((size_t)(b * C_IN + c)) * N_TOK + n];
}

// Prep: W[K][Ncol] f32 -> Wt[Ncol][K] bf16
__global__ void prep_wt(const float* __restrict__ w, bf16* __restrict__ wt,
                        int K, int Ncol) {
  int idx = blockIdx.x * 256 + threadIdx.x;
  if (idx >= K * Ncol) return;
  int col = idx % Ncol;
  int k   = idx / Ncol;
  wt[(size_t)col * K + k] = (bf16)w[idx];
}

// ---------------------------------------------------------------------------
// QKV projection: P = t[9216][256] @ Wt^T  (Wt is [512][256] bf16)
// One wave computes a 16x64 strip: A fragment loaded once per k-chunk,
// reused across 4 B tiles (4x less A traffic, 4 wmma per A load pair).
// mode 0: dst[b][h][n][d]   (Q, K)
// mode 1: dst[b][h][d][n]   (V transposed for PV wmma B-fragments)
// ---------------------------------------------------------------------------
__global__ void __launch_bounds__(256)
gemm_qkv(const bf16* __restrict__ A, const bf16* __restrict__ Bt,
         bf16* __restrict__ dst, int mode) {
  const int lane = threadIdx.x & 31;
  const int ln = lane & 15, hf = lane >> 4;
  const int gid = blockIdx.x * 8 + (threadIdx.x >> 5);  // 4608 strips exact
  const int tn0 = (gid & 7) * 4;  // 512/64 = 8 strip columns -> 4 n-tiles
  const int tm  = gid >> 3;       // 9216/16 = 576 row tiles

  v8f acc[4] = {};
#pragma unroll
  for (int k0 = 0; k0 < C_IN; k0 += 32) {
    v16bf a = load_frag(A, C_IN, tm * 16 + ln, k0, lane);
#pragma unroll
    for (int j = 0; j < 4; ++j) {
      v16bf b = load_frag(Bt, C_IN, (tn0 + j) * 16 + ln, k0, lane);
      acc[j] = wmma_bf16(a, b, acc[j]);
    }
  }

#pragma unroll
  for (int j = 0; j < 4; ++j) {
    const int col = (tn0 + j) * 16 + ln;
    const int h = col >> 6, d = col & 63;
#pragma unroll
    for (int r = 0; r < 8; ++r) {
      int M = tm * 16 + r + 8 * hf;
      int b_ = M / N_TOK, n = M % N_TOK;
      size_t o = (mode == 0)
        ? ((size_t)((b_ * NHEAD + h) * N_TOK + n)) * DHEAD + d
        : ((size_t)((b_ * NHEAD + h) * DHEAD + d)) * N_TOK + n;
      dst[o] = (bf16)acc[j][r];
    }
  }
}

// ---------------------------------------------------------------------------
// Flash attention, one wave per (b, h, 16-row Q tile).
// S = Q Kt (bf16 wmma, f32 acc) -> LDS -> online softmax (lanes 0..15 own
// rows) -> bf16 P in LDS -> A-fragment via ds_load_b128 -> O += P V.
// Post-softmax /sqrt(dk) quirk folded into final 1/(rowsum*8).
// ---------------------------------------------------------------------------
__global__ void __launch_bounds__(256)
flash_attn(const bf16* __restrict__ Q, const bf16* __restrict__ K,
           const bf16* __restrict__ VT, bf16* __restrict__ O) {
  __shared__ __align__(16) float Sh[8][16][32];
  __shared__ __align__(16) bf16  Ph[8][16][32];
  __shared__ float rmaxs[8][16], rsums[8][16], ralph[8][16];

  const int wv = threadIdx.x >> 5, lane = threadIdx.x & 31;
  const int ln = lane & 15, hf = lane >> 4;
  const int gw = blockIdx.x * 8 + wv;         // 4608 waves exact
  const int qt = gw % (N_TOK / 16);
  const int bh = gw / (N_TOK / 16);
  const int h = bh & (NHEAD - 1), b = bh >> 3;

  const bf16* Qh  = Q  + (size_t)bh * N_TOK * DHEAD;
  const bf16* Kh  = K  + (size_t)bh * N_TOK * DHEAD;
  const bf16* VTh = VT + (size_t)bh * DHEAD * N_TOK;

  // Q tile A-fragments for both 32-wide d chunks (reused across all m)
  v16bf aq0 = load_frag(Qh, DHEAD, qt * 16 + ln, 0,  lane);
  v16bf aq1 = load_frag(Qh, DHEAD, qt * 16 + ln, 32, lane);

  v8f o0 = {}, o1 = {}, o2 = {}, o3 = {};
  if (lane < 16) { rmaxs[wv][lane] = -INFINITY; rsums[wv][lane] = 0.0f; }
  // ordering vs. first softmax read is covered by the wait_lds() inside the
  // loop (same-wave LDS ops are tracked by DScnt)

  for (int m0 = 0; m0 < N_TOK; m0 += 32) {
    // ---- scores: two 16-col tiles ----
    v8f s0 = {}, s1 = {};
    v16bf bk;
    bk = load_frag(Kh, DHEAD, m0 + ln,      0,  lane); s0 = wmma_bf16(aq0, bk, s0);
    bk = load_frag(Kh, DHEAD, m0 + ln,      32, lane); s0 = wmma_bf16(aq1, bk, s0);
    bk = load_frag(Kh, DHEAD, m0 + 16 + ln, 0,  lane); s1 = wmma_bf16(aq0, bk, s1);
    bk = load_frag(Kh, DHEAD, m0 + 16 + ln, 32, lane); s1 = wmma_bf16(aq1, bk, s1);

#pragma unroll
    for (int r = 0; r < 8; ++r) {
      Sh[wv][r + 8 * hf][ln]      = s0[r];
      Sh[wv][r + 8 * hf][16 + ln] = s1[r];
    }
    wait_lds();

    // ---- online softmax: lane i owns row i ----
    if (lane < 16) {
      float mx = rmaxs[wv][lane];
      float cm = -INFINITY;
      float v[32];
#pragma unroll
      for (int j = 0; j < 32; ++j) { v[j] = Sh[wv][lane][j]; cm = fmaxf(cm, v[j]); }
      float nm = fmaxf(mx, cm);
      float alpha = __expf(mx - nm);
      float sum = 0.0f;
#pragma unroll
      for (int j = 0; j < 32; ++j) {
        float e = __expf(v[j] - nm);
        sum += e;
        Ph[wv][lane][j] = (bf16)e;
      }
      rsums[wv][lane] = rsums[wv][lane] * alpha + sum;
      rmaxs[wv][lane] = nm;
      ralph[wv][lane] = alpha;
    }
    wait_lds();

    // ---- rescale O accumulators by per-row alpha ----
#pragma unroll
    for (int r = 0; r < 8; ++r) {
      float ar = ralph[wv][r + 8 * hf];
      o0[r] *= ar; o1[r] *= ar; o2[r] *= ar; o3[r] *= ar;
    }

    // ---- O += P @ V : P from LDS, V^T b-fragments from global ----
    v16bf pa = load_frag(&Ph[wv][0][0], 32, ln, 0, lane);
    v16bf bv;
    bv = load_frag(VTh, N_TOK, 0  + ln, m0, lane); o0 = wmma_bf16(pa, bv, o0);
    bv = load_frag(VTh, N_TOK, 16 + ln, m0, lane); o1 = wmma_bf16(pa, bv, o1);
    bv = load_frag(VTh, N_TOK, 32 + ln, m0, lane); o2 = wmma_bf16(pa, bv, o2);
    bv = load_frag(VTh, N_TOK, 48 + ln, m0, lane); o3 = wmma_bf16(pa, bv, o3);
  }

  // ---- epilogue: O[b*N+n][h*64+d] = o / (rowsum * sqrt(64)) ----
  wait_lds();
#pragma unroll
  for (int r = 0; r < 8; ++r) {
    int row = r + 8 * hf;
    float inv = 1.0f / (rsums[wv][row] * 8.0f);
    int n = qt * 16 + row;
    size_t base = ((size_t)(b * N_TOK + n)) * QKVDIM + h * DHEAD + ln;
    O[base +  0] = (bf16)(o0[r] * inv);
    O[base + 16] = (bf16)(o1[r] * inv);
    O[base + 32] = (bf16)(o2[r] * inv);
    O[base + 48] = (bf16)(o3[r] * inv);
  }
}

// ---------------------------------------------------------------------------
// Output projection: out = O[9216][512] @ WoT^T (WoT = [256][512] bf16),
// 16x64 strip per wave, f32 epilogue scattered to [B][C][H*W].
// ---------------------------------------------------------------------------
__global__ void __launch_bounds__(256)
gemm_out(const bf16* __restrict__ A, const bf16* __restrict__ Bt,
         float* __restrict__ out) {
  const int lane = threadIdx.x & 31;
  const int ln = lane & 15, hf = lane >> 4;
  const int gid = blockIdx.x * 8 + (threadIdx.x >> 5);  // 2304 strips exact
  const int tn0 = (gid & 3) * 4;  // 256/64 = 4 strip columns -> 4 n-tiles
  const int tm  = gid >> 2;       // 576 row tiles

  v8f acc[4] = {};
#pragma unroll
  for (int k0 = 0; k0 < QKVDIM; k0 += 32) {
    v16bf a = load_frag(A, QKVDIM, tm * 16 + ln, k0, lane);
#pragma unroll
    for (int j = 0; j < 4; ++j) {
      v16bf b = load_frag(Bt, QKVDIM, (tn0 + j) * 16 + ln, k0, lane);
      acc[j] = wmma_bf16(a, b, acc[j]);
    }
  }

#pragma unroll
  for (int j = 0; j < 4; ++j) {
    const int c = (tn0 + j) * 16 + ln;
#pragma unroll
    for (int r = 0; r < 8; ++r) {
      int M = tm * 16 + r + 8 * hf;
      int b_ = M / N_TOK, n = M % N_TOK;
      out[((size_t)(b_ * C_IN + c)) * N_TOK + n] = acc[j][r];
    }
  }
}

// ---------------------------------------------------------------------------
extern "C" void kernel_launch(void* const* d_in, const int* in_sizes, int n_in,
                              void* d_out, int out_size, void* d_ws, size_t ws_size,
                              hipStream_t stream) {
  const float* x  = (const float*)d_in[0];
  const float* Wq = (const float*)d_in[1];
  const float* Wk = (const float*)d_in[2];
  const float* Wv = (const float*)d_in[3];
  const float* Wo = (const float*)d_in[4];
  float* out = (float*)d_out;

  bf16* w = (bf16*)d_ws;
  size_t off = 0;
  bf16* tB  = w + off; off += (size_t)BATCH * N_TOK * C_IN;      // 2359296
  bf16* WqT = w + off; off += (size_t)C_IN * QKVDIM;             // 131072
  bf16* WkT = w + off; off += (size_t)C_IN * QKVDIM;
  bf16* WvT = w + off; off += (size_t)C_IN * QKVDIM;
  bf16* WoT = w + off; off += (size_t)QKVDIM * C_IN;
  bf16* Qb  = w + off; off += (size_t)BATCH * N_TOK * QKVDIM;    // 4718592
  bf16* Kb  = w + off; off += (size_t)BATCH * N_TOK * QKVDIM;
  bf16* VTb = w + off; off += (size_t)BATCH * N_TOK * QKVDIM;
  bf16* Ob  = w + off; off += (size_t)BATCH * N_TOK * QKVDIM;

  // prep
  prep_tokens<<<(BATCH * N_TOK * C_IN) / 256, 256, 0, stream>>>(x, tB);
  prep_wt<<<(C_IN * QKVDIM + 255) / 256, 256, 0, stream>>>(Wq, WqT, C_IN, QKVDIM);
  prep_wt<<<(C_IN * QKVDIM + 255) / 256, 256, 0, stream>>>(Wk, WkT, C_IN, QKVDIM);
  prep_wt<<<(C_IN * QKVDIM + 255) / 256, 256, 0, stream>>>(Wv, WvT, C_IN, QKVDIM);
  prep_wt<<<(QKVDIM * C_IN + 255) / 256, 256, 0, stream>>>(Wo, WoT, QKVDIM, C_IN);

  // projections (4608 strips / 8 waves = 576 blocks)
  gemm_qkv<<<576, 256, 0, stream>>>(tB, WqT, Qb, 0);
  gemm_qkv<<<576, 256, 0, stream>>>(tB, WkT, Kb, 0);
  gemm_qkv<<<576, 256, 0, stream>>>(tB, WvT, VTb, 1);

  // attention (4608 waves / 8 = 576 blocks)
  flash_attn<<<576, 256, 0, stream>>>(Qb, Kb, VTb, Ob);

  // output projection (2304 strips / 8 = 288 blocks)
  gemm_out<<<288, 256, 0, stream>>>(Ob, WoT, out);
}
